// GroupedQueryAttention_34316788695052
// MI455X (gfx1250) — matrix operand order repaired and checked by reading, not compiled, not run on hardware
//
#include <hip/hip_runtime.h>
#include <hip/hip_bf16.h>

// ---------------------------------------------------------------------------
// GQA forward on gfx1250 (CDNA5, wave32):
//   QKV proj (bf16 WMMA, RoPE fused) -> flash attention (TDM-staged K/V in
//   LDS, bf16 WMMA QK^T and PV, fp32 online softmax) -> O proj (WMMA, fp32).
// ---------------------------------------------------------------------------

typedef __attribute__((ext_vector_type(16))) __bf16 v16bf;
typedef __attribute__((ext_vector_type(8)))  float  v8f;
typedef __attribute__((ext_vector_type(4)))  unsigned int v4u;
typedef __attribute__((ext_vector_type(8)))  int   v8i;
typedef __attribute__((ext_vector_type(4)))  int   v4i;

union AOp {
  v16bf v;
  uint4 u[2];
};

__device__ __forceinline__ unsigned short f2bf(float f) {
  unsigned int u = __float_as_uint(f);
  u += 0x7FFFu + ((u >> 16) & 1u);   // round-to-nearest-even
  return (unsigned short)(u >> 16);
}

__global__ void __launch_bounds__(256)
cvt_f32_bf16(const float* __restrict__ in, unsigned short* __restrict__ out, int n) {
  int i = blockIdx.x * blockDim.x + threadIdx.x;
  if (i < n) out[i] = f2bf(in[i]);
}

// --------------------------- TDM helpers -----------------------------------
// 2D tensor_load_to_lds: tile (tile0 x tile1) of 2-byte elements, row stride
// stride0 (elements), from gptr into LDS offset lds_off.  D# per ISA §8.
__device__ __forceinline__ void tdm_load_2d(const unsigned short* gptr,
                                            unsigned lds_off,
                                            unsigned tile0, unsigned tile1,
                                            unsigned long long stride0,
                                            unsigned tdim0, unsigned tdim1) {
#if defined(__gfx1250__)
  unsigned long long ga = (unsigned long long)(uintptr_t)gptr;
  v4u g0;
  g0[0] = 1u;                                    // count=1, user descriptor
  g0[1] = lds_off;                               // lds_addr (bytes)
  g0[2] = (unsigned)(ga & 0xFFFFFFFFu);          // global_addr[31:0]
  g0[3] = (unsigned)((ga >> 32) & 0x01FFFFFFu) | (2u << 30);  // [56:32] | type=2
  v8i g1;
  g1[0] = (int)(1u << 16);                       // data_size=1 (2 bytes)
  g1[1] = (int)((tdim0 & 0xFFFFu) << 16);        // tensor_dim0[15:0] @ bits63:48
  g1[2] = (int)((tdim0 >> 16) | ((tdim1 & 0xFFFFu) << 16));
  g1[3] = (int)((tdim1 >> 16) | (tile0 << 16));  // tensor_dim1 hi | tile_dim0
  g1[4] = (int)(tile1 & 0xFFFFu);                // tile_dim1 (tile_dim2 = 0)
  g1[5] = (int)(unsigned)(stride0 & 0xFFFFFFFFu);        // dim0_stride[31:0]
  g1[6] = (int)((unsigned)(stride0 >> 32) & 0xFFFFu);    // dim0_stride[47:32]
  g1[7] = 0;
  v4i z4 = {0, 0, 0, 0};
#if defined(__clang_major__) && __clang_major__ >= 23
  v8i z8 = {0, 0, 0, 0, 0, 0, 0, 0};
  __builtin_amdgcn_tensor_load_to_lds(g0, g1, z4, z4, z8, 0);
#else
  __builtin_amdgcn_tensor_load_to_lds(g0, g1, z4, z4, 0);
#endif
#else
  // Host-pass placeholder (never executed on device): plain copy.
  unsigned short* l = (unsigned short*)(uintptr_t)lds_off;
  for (unsigned r = 0; r < tile1; ++r)
    for (unsigned c = 0; c < tile0; ++c)
      l[r * tile0 + c] = gptr[r * stride0 + c];
  (void)tdim0; (void)tdim1;
#endif
}

__device__ __forceinline__ void tdm_wait0() {
#if defined(__gfx1250__)
  __builtin_amdgcn_s_wait_tensorcnt(0);
#else
  __syncthreads();
#endif
}

__device__ __forceinline__ unsigned lds_off_of(const void* p) {
  return (unsigned)(uintptr_t)p;   // LDS aperture: addr[31:0] = LDS byte offset
}

// ------------------------------- GEMM --------------------------------------
// D = A[M,K] * W[N,K]^T + bias.  M = 4096 (B*S), K = 1024.
// One wave computes a 32x64 output tile (B operand reused across 2 m-subtiles).
// grid = (M/32, N/64), block = 32.
// mode 0: Q proj, +RoPE, store bf16 [b][16][2048][64]
// mode 1: K proj, +RoPE, store bf16 [b][4][2048][64]
// mode 2: V proj, store TRANSPOSED bf16 [b][4][64][2048]
// mode 3: O proj, store fp32 [4096][1024]
__global__ void __launch_bounds__(32)
wmma_gemm(const unsigned short* __restrict__ A,
          const unsigned short* __restrict__ W,
          const float* __restrict__ bias,
          void* __restrict__ out, int mode) {
  const int lane = threadIdx.x & 31;
  const int hf   = lane >> 4;
  const int mn   = lane & 15;
  const int mtile = blockIdx.x;   // 32 rows
  const int ntile = blockIdx.y;   // 64 cols

  const unsigned short* arow0 = A + (size_t)(mtile * 32 + mn) * 1024;
  const unsigned short* arow1 = arow0 + 16 * 1024;
  v8f acc[2][4] = {};

  for (int k0 = 0; k0 < 1024; k0 += 32) {
    if (k0 + 64 < 1024) {
      __builtin_prefetch(arow0 + k0 + 64, 0, 0);
      __builtin_prefetch(arow1 + k0 + 64, 0, 0);
    }
    AOp a0, a1;  // A 16x32: lane=m; vgpr i -> k = (i>>2)*16 + hf*8 + (i&3)*2
    a0.u[0] = *reinterpret_cast<const uint4*>(arow0 + k0 + hf * 8);
    a0.u[1] = *reinterpret_cast<const uint4*>(arow0 + k0 + 16 + hf * 8);
    a1.u[0] = *reinterpret_cast<const uint4*>(arow1 + k0 + hf * 8);
    a1.u[1] = *reinterpret_cast<const uint4*>(arow1 + k0 + 16 + hf * 8);
#pragma unroll
    for (int t = 0; t < 4; ++t) {
      const unsigned short* wrow =
          W + (size_t)(ntile * 64 + t * 16 + mn) * 1024 + k0;
      AOp b;  // B 32x16: lane=n; vgpr i -> k = (i>>2)*16 + hf*8 + (i&3)*2
      b.u[0] = *reinterpret_cast<const uint4*>(wrow + hf * 8);
      b.u[1] = *reinterpret_cast<const uint4*>(wrow + 16 + hf * 8);
      acc[0][t] = __builtin_amdgcn_wmma_f32_16x16x32_bf16(
          false, a0.v, false, b.v, (short)0, acc[0][t], false, false);
      acc[1][t] = __builtin_amdgcn_wmma_f32_16x16x32_bf16(
          false, a1.v, false, b.v, (short)0, acc[1][t], false, false);
    }
  }

  // Epilogue. C/D layout: lane=n, row m = i + 8*hf.
#pragma unroll
  for (int sub = 0; sub < 2; ++sub) {
#pragma unroll
    for (int t = 0; t < 4; ++t) {
      const int col  = ntile * 64 + t * 16 + mn;
      const float bc = bias[col];
#pragma unroll
      for (int i = 0; i < 8; ++i) {
        const int r = mtile * 32 + sub * 16 + i + hf * 8;  // row in [0,4096)
        const int b = r >> 11;
        const int s = r & 2047;
        float val = acc[sub][t][i] + bc;
        if (mode == 0 || mode == 1) {
          // RoPE, faithful to reference quirks:
          //   theta = 1/(10000^(2u)/64), fp32 overflow -> theta == 0
          //   out_even = X1*c - X2*s ; out_odd = X1*c + X2*s
          const int d = col & 63;
          const float partner = __shfl_xor(val, 1, 32);
          const float X1 = (d & 1) ? partner : val;
          const float X2 = (d & 1) ? val : partner;
          const float pw    = powf(10000.0f, (float)(d & ~1));
          const float theta = 1.0f / (pw / 64.0f);
          const float ang   = (float)s * theta;
          const float c = cosf(ang), sn = sinf(ang);
          val = (d & 1) ? (X1 * c + X2 * sn) : (X1 * c - X2 * sn);
          unsigned short* o = (unsigned short*)out;
          const int hd = col >> 6;
          if (mode == 0)
            o[(((size_t)(b * 16 + hd) * 2048 + s) << 6) + d] = f2bf(val);
          else
            o[(((size_t)(b * 4 + hd) * 2048 + s) << 6) + d] = f2bf(val);
        } else if (mode == 2) {
          const int d = col & 63, hd = col >> 6;
          ((unsigned short*)out)[((size_t)((b * 4 + hd) * 64 + d) << 11) + s] =
              f2bf(val);
        } else {
          ((float*)out)[(size_t)r * 1024 + col] = val;
        }
      }
    }
  }
}

// --------------------------- Flash attention --------------------------------
// grid = (B*16 heads, S/16 q-tiles), block = 32 (1 wave).
// K/V blocks are staged into LDS by the Tensor Data Mover, double-buffered.
__global__ void __launch_bounds__(32)
flash_attn(const unsigned short* __restrict__ Q,
           const unsigned short* __restrict__ K,
           const unsigned short* __restrict__ Vt,
           unsigned short* __restrict__ att) {
  const int lane = threadIdx.x & 31;
  const int hf = lane >> 4;
  const int mn = lane & 15;
  const int bh = blockIdx.x;            // b*16 + h
  const int b = bh >> 4, h = bh & 15, hkv = h >> 2;
  const int q0 = blockIdx.y * 16;

  const unsigned short* Qb = Q + (((size_t)bh * 2048 + q0) << 6);
  const unsigned short* Kb = K + (((size_t)(b * 4 + hkv) * 2048) << 6);
  const unsigned short* Vb = Vt + (((size_t)(b * 4 + hkv) * 64) << 11);

  __shared__ __align__(16) unsigned short Kl[2][32 * 64];  // 32 kv rows x 64 d
  __shared__ __align__(16) unsigned short Vl[2][64 * 32];  // 64 d rows x 32 kv
  __shared__ __align__(16) unsigned short Pl[16 * 32];

  // Q tile as two A-operands (d 0..31, 32..63), resident for whole kv loop.
  AOp qlo, qhi;
  {
    const unsigned short* qrow = Qb + mn * 64;
    qlo.u[0] = *reinterpret_cast<const uint4*>(qrow + hf * 8);
    qlo.u[1] = *reinterpret_cast<const uint4*>(qrow + 16 + hf * 8);
    qhi.u[0] = *reinterpret_cast<const uint4*>(qrow + 32 + hf * 8);
    qhi.u[1] = *reinterpret_cast<const uint4*>(qrow + 48 + hf * 8);
  }

  const float NEG = -__builtin_inff();
  float mrow[8], lrow[8];
#pragma unroll
  for (int i = 0; i < 8; ++i) { mrow[i] = NEG; lrow[i] = 0.0f; }
  v8f oc[4] = {};

  const int nblk = (q0 + 15) / 32 + 1;

  // Prologue: TDM-stage block 0.
  tdm_load_2d(Kb, lds_off_of(&Kl[0][0]), 64, 32, 64, 64, 2048);
  tdm_load_2d(Vb, lds_off_of(&Vl[0][0]), 32, 64, 2048, 2048, 64);
  tdm_wait0();

  for (int j = 0; j < nblk; ++j) {
    const int k0 = j * 32;
    // Kick off TDM for the next block while we compute this one.
    if (j + 1 < nblk) {
      const int nb = (j + 1) & 1;
      tdm_load_2d(Kb + (size_t)(k0 + 32) * 64, lds_off_of(&Kl[nb][0]),
                  64, 32, 64, 64, 2048);
      tdm_load_2d(Vb + (k0 + 32), lds_off_of(&Vl[nb][0]),
                  32, 64, 2048, 2048, 64);
    }
    const unsigned short* Kc = &Kl[j & 1][0];
    const unsigned short* Vc = &Vl[j & 1][0];

    // S = Q * K^T for this 16x32 kv block (2 tiles x 2 K-steps of 32).
    v8f st[2];
#pragma unroll
    for (int t = 0; t < 2; ++t) {
      const unsigned short* krow = Kc + (t * 16 + mn) * 64;
      AOp b0, b1;  // B operand: lane = kv column n, vgpr i -> d = (i>>2)*16 + hf*8 + (i&3)*2
      b0.u[0] = *reinterpret_cast<const uint4*>(krow + hf * 8);
      b0.u[1] = *reinterpret_cast<const uint4*>(krow + 16 + hf * 8);
      b1.u[0] = *reinterpret_cast<const uint4*>(krow + 32 + hf * 8);
      b1.u[1] = *reinterpret_cast<const uint4*>(krow + 48 + hf * 8);
      v8f z = {};
      z = __builtin_amdgcn_wmma_f32_16x16x32_bf16(false, qlo.v, false, b0.v,
                                                  (short)0, z, false, false);
      z = __builtin_amdgcn_wmma_f32_16x16x32_bf16(false, qhi.v, false, b1.v,
                                                  (short)0, z, false, false);
      st[t] = z;
    }

    // Scale, causal mask, fp32 online softmax; write P (bf16) to LDS.
#pragma unroll
    for (int i = 0; i < 8; ++i) {
      const int row = q0 + i + hf * 8;
      float s0 = st[0][i] * 0.125f;              // 1/sqrt(64)
      float s1 = st[1][i] * 0.125f;
      if (k0 + mn > row)      s0 = NEG;
      if (k0 + 16 + mn > row) s1 = NEG;
      float rm = fmaxf(s0, s1);
#pragma unroll
      for (int off = 1; off < 16; off <<= 1) rm = fmaxf(rm, __shfl_xor(rm, off, 32));
      const float nm   = fmaxf(mrow[i], rm);
      const float corr = expf(mrow[i] - nm);
      const float p0 = expf(s0 - nm);
      const float p1 = expf(s1 - nm);
      Pl[(i + hf * 8) * 32 + mn]      = f2bf(p0);
      Pl[(i + hf * 8) * 32 + 16 + mn] = f2bf(p1);
      float rs = p0 + p1;
#pragma unroll
      for (int off = 1; off < 16; off <<= 1) rs += __shfl_xor(rs, off, 32);
      lrow[i] = lrow[i] * corr + rs;
      mrow[i] = nm;
#pragma unroll
      for (int nt = 0; nt < 4; ++nt) oc[nt][i] *= corr;
    }
    __syncthreads();

    // Reload P as A-operand (row = lane) from LDS.
    AOp pa;
    {
      const unsigned short* prow = &Pl[mn * 32];
      pa.u[0] = *reinterpret_cast<const uint4*>(prow + hf * 8);
      pa.u[1] = *reinterpret_cast<const uint4*>(prow + 16 + hf * 8);
    }
    __syncthreads();

    // O += P * V from the TDM-staged transposed-V block.
#pragma unroll
    for (int nt = 0; nt < 4; ++nt) {
      const unsigned short* vrow = Vc + (nt * 16 + mn) * 32;
      AOp bv;
      bv.u[0] = *reinterpret_cast<const uint4*>(vrow + hf * 8);
      bv.u[1] = *reinterpret_cast<const uint4*>(vrow + 16 + hf * 8);
      oc[nt] = __builtin_amdgcn_wmma_f32_16x16x32_bf16(
          false, pa.v, false, bv.v, (short)0, oc[nt], false, false);
    }

    tdm_wait0();   // next iteration's buffers are ready past this point
  }

  // Normalize and store attention output bf16 [b][s][h*64 + d].
#pragma unroll
  for (int nt = 0; nt < 4; ++nt) {
#pragma unroll
    for (int i = 0; i < 8; ++i) {
      const int s = q0 + i + hf * 8;
      const int d = h * 64 + nt * 16 + mn;
      att[((size_t)(b * 2048 + s) << 10) + d] = f2bf(oc[nt][i] / lrow[i]);
    }
  }
}

extern "C" void kernel_launch(void* const* d_in, const int* in_sizes, int n_in,
                              void* d_out, int out_size, void* d_ws, size_t ws_size,
                              hipStream_t stream) {
  (void)in_sizes; (void)n_in; (void)out_size; (void)ws_size;
  const float* x  = (const float*)d_in[0];
  const float* wq = (const float*)d_in[1];
  const float* bq = (const float*)d_in[2];
  const float* wk = (const float*)d_in[3];
  const float* bk = (const float*)d_in[4];
  const float* wv = (const float*)d_in[5];
  const float* bv = (const float*)d_in[6];
  const float* wo = (const float*)d_in[7];
  const float* bo = (const float*)d_in[8];

  char* ws = (char*)d_ws;
  size_t off = 0;
  auto carve = [&](size_t bytes) { void* p = ws + off; off += bytes; return p; };
  unsigned short* xb  = (unsigned short*)carve(4096ull * 1024 * 2);
  unsigned short* wqb = (unsigned short*)carve(1024ull * 1024 * 2);
  unsigned short* wkb = (unsigned short*)carve(256ull * 1024 * 2);
  unsigned short* wvb = (unsigned short*)carve(256ull * 1024 * 2);
  unsigned short* wob = (unsigned short*)carve(1024ull * 1024 * 2);
  unsigned short* Qb  = (unsigned short*)carve(2ull * 16 * 2048 * 64 * 2);
  unsigned short* Kb  = (unsigned short*)carve(2ull * 4 * 2048 * 64 * 2);
  unsigned short* Vt  = (unsigned short*)carve(2ull * 4 * 2048 * 64 * 2);
  unsigned short* att = (unsigned short*)carve(4096ull * 1024 * 2);

  auto cvt = [&](const float* in, unsigned short* out, int n) {
    cvt_f32_bf16<<<(n + 255) / 256, 256, 0, stream>>>(in, out, n);
  };
  cvt(x,  xb,  4096 * 1024);
  cvt(wq, wqb, 1024 * 1024);
  cvt(wk, wkb, 256 * 1024);
  cvt(wv, wvb, 256 * 1024);
  cvt(wo, wob, 1024 * 1024);

  wmma_gemm<<<dim3(128, 16), 32, 0, stream>>>(xb, wqb, bq, Qb, 0);
  wmma_gemm<<<dim3(128, 4),  32, 0, stream>>>(xb, wkb, bk, Kb, 1);
  wmma_gemm<<<dim3(128, 4),  32, 0, stream>>>(xb, wvb, bv, Vt, 2);
  flash_attn<<<dim3(32, 128), 32, 0, stream>>>(Qb, Kb, Vt, att);
  wmma_gemm<<<dim3(128, 16), 32, 0, stream>>>(att, wob, bo, d_out, 3);
}
